// MapBasedMultiHeadAttention_63264868270464
// MI455X (gfx1250) — matrix operand order repaired and checked by reading, not compiled, not run on hardware
//
#include <hip/hip_runtime.h>
#include <hip/hip_bf16.h>
#include <math.h>

// Shapes (compile-time): B=8, L=1024, DM=512, H=8, DK=DV=64
#define Bn 8
#define Ln 1024
#define DMn 512
#define Hn 8
#define DKn 64
#define DVn 64

typedef __attribute__((ext_vector_type(16))) _Float16 v16h;
typedef __attribute__((ext_vector_type(8)))  float    v8f;
typedef __attribute__((ext_vector_type(4)))  float    f32x4;

// ---------------------------------------------------------------------------
// 1) Fold Wk (per-head) with Wmap[DK:] : Weff[d][h] = sum_j Wk[d, h*64+j]*Wmap[64+j]
// ---------------------------------------------------------------------------
__global__ __launch_bounds__(256) void weff_kernel(const float* __restrict__ Wk,
                                                   const float* __restrict__ Wmap,
                                                   float* __restrict__ weff) {
    int i = blockIdx.x * 256 + threadIdx.x;   // 0..4095  (512 d * 8 h)
    int d = i >> 3, h = i & 7;
    float acc = 0.f;
    #pragma unroll 8
    for (int j = 0; j < 64; ++j)
        acc += Wk[d * DMn + h * 64 + j] * Wmap[DKn + j];
    weff[d * 8 + h] = acc;
}

// ---------------------------------------------------------------------------
// 2) sk[h,b,l] = k[b,l,:] . Weff[:,h]  ;  p[h,b,:] = softmax_l(sk[h,b,:])
//    One 1024-thread block per batch b. LDS: 16K (Weff) + 32K (sk) + 4K (red).
// ---------------------------------------------------------------------------
__global__ __launch_bounds__(1024) void sk_softmax_kernel(const float* __restrict__ k,
                                                          const float* __restrict__ weff,
                                                          float* __restrict__ p) {
    __shared__ float w_lds[DMn * 8];
    __shared__ float sk_lds[8 * Ln];
    __shared__ float red[Ln];
    const int b = blockIdx.x;
    const int t = threadIdx.x;               // == l

    for (int i = t; i < DMn * 8; i += 1024) w_lds[i] = weff[i];
    __syncthreads();

    float acc[8] = {0.f, 0.f, 0.f, 0.f, 0.f, 0.f, 0.f, 0.f};
    const float* krow = k + ((size_t)b * Ln + t) * DMn;
    for (int d = 0; d < DMn; d += 4) {
        f32x4 kv = *(const f32x4*)(krow + d);
        #pragma unroll
        for (int h = 0; h < 8; ++h) {
            acc[h] += kv.x * w_lds[(d + 0) * 8 + h];
            acc[h] += kv.y * w_lds[(d + 1) * 8 + h];
            acc[h] += kv.z * w_lds[(d + 2) * 8 + h];
            acc[h] += kv.w * w_lds[(d + 3) * 8 + h];
        }
    }
    #pragma unroll
    for (int h = 0; h < 8; ++h) sk_lds[h * Ln + t] = acc[h];
    __syncthreads();

    for (int h = 0; h < 8; ++h) {
        float vsk = sk_lds[h * Ln + t];
        red[t] = vsk; __syncthreads();
        for (int s = 512; s > 0; s >>= 1) {
            if (t < s) red[t] = fmaxf(red[t], red[t + s]);
            __syncthreads();
        }
        float m = red[0]; __syncthreads();
        float e = __expf(vsk - m);
        red[t] = e; __syncthreads();
        for (int s = 512; s > 0; s >>= 1) {
            if (t < s) red[t] += red[t + s];
            __syncthreads();
        }
        float denom = red[0]; __syncthreads();
        p[((size_t)h * Bn + b) * Ln + t] = e / denom;  // exact softmax row
    }
}

// ---------------------------------------------------------------------------
// 3) attn broadcast: attn[(h*B+b), q, :] = p[h,b,:] for all q.
//    Dominant cost (268 MB). Row cached in registers, nontemporal b128 stores.
//    grid = (64 hb, 16 chunks), block = 256; each block streams 64 q-rows.
// ---------------------------------------------------------------------------
__global__ __launch_bounds__(256) void attn_bcast_kernel(const float* __restrict__ p,
                                                         float* __restrict__ attn) {
    const int hb = blockIdx.x;       // 0..63
    const int chunk = blockIdx.y;    // 0..15
    f32x4 val = *(const f32x4*)(p + (size_t)hb * Ln + threadIdx.x * 4);
    float* base = attn + ((size_t)hb * Ln + (size_t)chunk * 64) * Ln + threadIdx.x * 4;
    #pragma unroll 4
    for (int r = 0; r < 64; ++r) {
        __builtin_nontemporal_store(val, (f32x4*)(base + (size_t)r * Ln));
    }
}

// ---------------------------------------------------------------------------
// 4) u[h,b,:] = sum_l p[h,b,l] * v[b,l,:]   via WMMA f32_16x16x32_f16.
//    Per batch b: A = P (16x1024, heads 8..15 zero-padded), Bmat = V (1024x512).
//    One wave per (b, 16-col tile): 32 tiles/b * 8 b = 256 waves = 32 blocks.
//    Fragments packed in contiguous k order (K = 16*lg + e) for both A and B.
// ---------------------------------------------------------------------------
__global__ __launch_bounds__(256) void pv_wmma_kernel(const float* __restrict__ p,
                                                      const float* __restrict__ v,
                                                      float* __restrict__ u) {
    const int wave = threadIdx.x >> 5;
    const int lane = threadIdx.x & 31;
    const int idx  = blockIdx.x * 8 + wave;   // 0..255
    const int b    = idx >> 5;                // 0..7
    const int nt   = idx & 31;                // 0..31 column tile
    const int n0   = nt * 16;
    const int lg   = lane >> 4;               // lane group (0: lanes 0-15, 1: 16-31)
    const int lm   = lane & 15;

    v8f c = {};
    for (int kt = 0; kt < 32; ++kt) {
        const int k0 = kt * 32;
        v16h a, bm;
        #pragma unroll
        for (int e = 0; e < 16; ++e) {
            // A 16x32 f16 layout: lane m=lm; K = e + lg*16
            int Ka = e + (lg << 4);
            float av = (lm < Hn) ? p[((size_t)lm * Bn + b) * Ln + k0 + Ka] : 0.0f;
            a[e] = (_Float16)av;
            // B 32x16 f16 layout: N = lm; K = e + lg*16
            int Kb = e + (lg << 4);
            bm[e] = (_Float16)v[((size_t)b * Ln + k0 + Kb) * DMn + n0 + lm];
        }
        c = __builtin_amdgcn_wmma_f32_16x16x32_f16(false, a, false, bm,
                                                   (short)0, c, false, false);
    }
    // C/D layout: lane group 0 holds M = r (0..7) == head; N = lm
    if (lg == 0) {
        #pragma unroll
        for (int r = 0; r < 8; ++r)
            u[((size_t)r * Bn + b) * DMn + n0 + lm] = c[r];
    }
}

// ---------------------------------------------------------------------------
// 5) ctx_flat[b, h*64+dv] = bv[h*64+dv] + u[h,b,:] . Wv[:, h*64+dv]
// ---------------------------------------------------------------------------
__global__ __launch_bounds__(64) void ctx_kernel(const float* __restrict__ u,
                                                 const float* __restrict__ Wv,
                                                 const float* __restrict__ bv,
                                                 float* __restrict__ ctx) {
    const int hb = blockIdx.x;           // h*8+b
    const int h = hb >> 3, b = hb & 7;
    const int dv = threadIdx.x;
    float acc = bv[h * 64 + dv];
    const float* ur = u + (size_t)hb * DMn;
    for (int d = 0; d < DMn; ++d)
        acc += ur[d] * Wv[(size_t)d * DMn + h * 64 + dv];
    ctx[b * DMn + h * 64 + dv] = acc;
}

// ---------------------------------------------------------------------------
// 6) fcout[b,:] = ctx[b,:] @ Wfc + bfc     (8 x 512 x 512, tiny)
// ---------------------------------------------------------------------------
__global__ __launch_bounds__(512) void fc_kernel(const float* __restrict__ ctx,
                                                 const float* __restrict__ Wfc,
                                                 const float* __restrict__ bfc,
                                                 float* __restrict__ fcout) {
    const int b = blockIdx.x, n = threadIdx.x;
    float acc = bfc[n];
    for (int m = 0; m < DMn; ++m)
        acc += ctx[b * DMn + m] * Wfc[(size_t)m * DMn + n];
    fcout[b * DMn + n] = acc;
}

// ---------------------------------------------------------------------------
// 7) out[b,l,:] = LayerNorm(fcout[b,:] + q[b,l,:]) * scale + bias
// ---------------------------------------------------------------------------
__global__ __launch_bounds__(256) void ln_kernel(const float* __restrict__ q,
                                                 const float* __restrict__ fcout,
                                                 const float* __restrict__ ls,
                                                 const float* __restrict__ lb,
                                                 float* __restrict__ out) {
    __shared__ float r1[256], r2[256];
    const int row = blockIdx.x;            // 0..8191
    const int b = row >> 10;
    const int t = threadIdx.x;
    const float* qr = q + (size_t)row * DMn;
    const float* fr = fcout + b * DMn;
    float v0 = qr[t] + fr[t];
    float v1 = qr[t + 256] + fr[t + 256];
    r1[t] = v0 + v1;
    r2[t] = v0 * v0 + v1 * v1;
    __syncthreads();
    for (int s = 128; s > 0; s >>= 1) {
        if (t < s) { r1[t] += r1[t + s]; r2[t] += r2[t + s]; }
        __syncthreads();
    }
    float mu  = r1[0] * (1.0f / DMn);
    float var = r2[0] * (1.0f / DMn) - mu * mu;
    float rstd = rsqrtf(var + 1e-5f);
    float* orow = out + (size_t)row * DMn;
    orow[t]       = (v0 - mu) * rstd * ls[t]       + lb[t];
    orow[t + 256] = (v1 - mu) * rstd * ls[t + 256] + lb[t + 256];
}

// ---------------------------------------------------------------------------
extern "C" void kernel_launch(void* const* d_in, const int* in_sizes, int n_in,
                              void* d_out, int out_size, void* d_ws, size_t ws_size,
                              hipStream_t stream) {
    const float* q    = (const float*)d_in[0];
    const float* k    = (const float*)d_in[1];
    const float* v    = (const float*)d_in[2];
    // d_in[3]=Wq, d_in[4]=bq : mathematically dead (softmax over k cancels sq)
    const float* Wk   = (const float*)d_in[5];
    // d_in[6]=bk : cancels in softmax
    const float* Wv   = (const float*)d_in[7];
    const float* bv   = (const float*)d_in[8];
    const float* Wmap = (const float*)d_in[9];
    // d_in[10]=bmap : cancels in softmax
    const float* Wfc  = (const float*)d_in[11];
    const float* bfc  = (const float*)d_in[12];
    const float* lns  = (const float*)d_in[13];
    const float* lnb  = (const float*)d_in[14];

    float* out  = (float*)d_out;                            // (B,L,DM) = 4,194,304 f32
    float* attn = out + (size_t)Bn * Ln * DMn;              // (H*B,L,L) = 67,108,864 f32

    float* ws    = (float*)d_ws;
    float* weff  = ws;                      // 512*8   = 4096 f32
    float* p     = weff + DMn * 8;          // 8*8*1024 = 65536 f32
    float* u     = p + Hn * Bn * Ln;        // 8*8*512  = 32768 f32
    float* ctx   = u + Hn * Bn * DMn;       // 8*512    = 4096 f32
    float* fcout = ctx + Bn * DMn;          // 8*512    = 4096 f32

    weff_kernel<<<16, 256, 0, stream>>>(Wk, Wmap, weff);
    sk_softmax_kernel<<<Bn, 1024, 0, stream>>>(k, weff, p);
    attn_bcast_kernel<<<dim3(Hn * Bn, 16), 256, 0, stream>>>(p, attn);
    pv_wmma_kernel<<<32, 256, 0, stream>>>(p, v, u);
    ctx_kernel<<<Hn * Bn, 64, 0, stream>>>(u, Wv, bv, ctx);
    fc_kernel<<<Bn, 512, 0, stream>>>(ctx, Wfc, bfc, fcout);
    ln_kernel<<<Bn * Ln, 256, 0, stream>>>(q, fcout, lns, lnb, out);
}
